// FPN_AAR_74028056314167
// MI455X (gfx1250) — matrix operand order repaired and checked by reading, not compiled, not run on hardware
//
#include <hip/hip_runtime.h>
#include <hip/hip_bf16.h>

typedef __attribute__((ext_vector_type(16))) __bf16 v16bf;
typedef __attribute__((ext_vector_type(8)))  __bf16 v8bf;
typedef __attribute__((ext_vector_type(8)))  float  v8f;

#define SHUF16(a,b) __builtin_shufflevector(a,b,0,1,2,3,4,5,6,7,8,9,10,11,12,13,14,15)
#define WMMA_BF16(A,B,C) __builtin_amdgcn_wmma_f32_16x16x32_bf16(false, A, false, B, (short)0, C, false, false)

__constant__ int c_rot[5][9] = {
  {0,1,2,3,4,5,6,7,8},
  {3,0,1,6,4,2,7,8,5},
  {6,3,0,7,4,1,8,5,2},
  {7,6,3,8,4,0,5,2,1},
  {8,7,6,5,4,3,2,1,0},
};

// ---------- NCHW f32 -> pixel-major [p][c] bf16, LDS-tiled transpose ----------
__global__ void k_transpose_bf16(const float* __restrict__ x, __bf16* __restrict__ xb,
                                 int C, int HW) {
  __shared__ __bf16 tile[32][33];
  const int pt = blockIdx.x * 32;
  const int ct = blockIdx.y * 32;
  const int tx = threadIdx.x & 31;
  const int ty = threadIdx.x >> 5;
#pragma unroll
  for (int r = 0; r < 32; r += 8)
    tile[ty + r][tx] = (__bf16)x[(size_t)(ct + ty + r) * HW + pt + tx];
  __syncthreads();
#pragma unroll
  for (int r = 0; r < 32; r += 8)
    xb[(size_t)(pt + ty + r) * C + ct + tx] = tile[tx][ty + r];
}

// ---------- same-layout f32 -> bf16 ----------
__global__ void k_f32_to_bf16(const float* __restrict__ src, __bf16* __restrict__ dst, size_t n) {
  for (size_t i = blockIdx.x * (size_t)blockDim.x + threadIdx.x; i < n;
       i += (size_t)gridDim.x * blockDim.x)
    dst[i] = (__bf16)src[i];
}

// ---------- build 5 rotated 3x3 kernels, bf16, layout [lev][r][o][t*256+i] ----------
__global__ void k_rotate_w(const float* __restrict__ w, __bf16* __restrict__ rw) {
  size_t n = 4ull * 5 * 256 * 9 * 256;
  for (size_t i = blockIdx.x * (size_t)blockDim.x + threadIdx.x; i < n;
       i += (size_t)gridDim.x * blockDim.x) {
    size_t tmp = i;
    int ich = (int)(tmp % 256); tmp /= 256;
    int t   = (int)(tmp % 9);   tmp /= 9;
    int o   = (int)(tmp % 256); tmp /= 256;
    int r   = (int)(tmp % 5);   tmp /= 5;
    int lev = (int)tmp;
    int src = c_rot[r][t];
    float val = w[(((size_t)lev * 256 + o) * 256 + ich) * 9 + src];
    rw[(((size_t)lev * 5 + r) * 256 + o) * 2304 + (size_t)t * 256 + ich] = (__bf16)val;
  }
}

// ---------- fold conv-bias + BN into per-channel scale/shift ----------
// sc = g*rsqrt(v+eps); sh = (cb - m)*sc + b   (indexed [lev][branch][c], 4*5*256)
__global__ void k_bnprep(const float* __restrict__ g, const float* __restrict__ b,
                         const float* __restrict__ m, const float* __restrict__ v,
                         const float* __restrict__ cb, float* __restrict__ sc,
                         float* __restrict__ sh) {
  int i = blockIdx.x * blockDim.x + threadIdx.x;
  if (i >= 4 * 5 * 256) return;
  int c = i & 255;
  int lev = i / (5 * 256);
  float s = g[i] * rsqrtf(v[i] + 1e-5f);
  sc[i] = s;
  sh[i] = (cb[lev * 256 + c] - m[i]) * s + b[i];
}

// ---------- lateral 1x1 conv, WMMA GEMM, 4M x 2N register blocking per wave ----------
__global__ void k_gemm_lat(const __bf16* __restrict__ Wm, const __bf16* __restrict__ X,
                           const float* __restrict__ bias, float* __restrict__ D, int C) {
  const int lane = threadIdx.x & 31;
  const int half = lane >> 4;
  const int lr   = lane & 15;
  const int wave = threadIdx.x >> 5;
  const int n0 = blockIdx.x * 32;
  const int mb = wave * 64;
  v8f acc[4][2] = {};
  // hoisted bases: all inner addressing is base + immediate
  const __bf16* b0 = X + (size_t)(n0 + lr) * C + 8 * half;
  const __bf16* b1 = X + (size_t)(n0 + 16 + lr) * C + 8 * half;
  const __bf16* a0 = Wm + (size_t)(mb + lr) * C + 8 * half;
  const __bf16* a1 = a0 + (size_t)16 * C;
  const __bf16* a2 = a0 + (size_t)32 * C;
  const __bf16* a3 = a0 + (size_t)48 * C;
  for (int k0 = 0; k0 < C; k0 += 32) {
    v16bf B0 = SHUF16(*(const v8bf*)(b0 + k0), *(const v8bf*)(b0 + k0 + 16));
    v16bf B1 = SHUF16(*(const v8bf*)(b1 + k0), *(const v8bf*)(b1 + k0 + 16));
    v16bf A0 = SHUF16(*(const v8bf*)(a0 + k0), *(const v8bf*)(a0 + k0 + 16));
    v16bf A1 = SHUF16(*(const v8bf*)(a1 + k0), *(const v8bf*)(a1 + k0 + 16));
    v16bf A2 = SHUF16(*(const v8bf*)(a2 + k0), *(const v8bf*)(a2 + k0 + 16));
    v16bf A3 = SHUF16(*(const v8bf*)(a3 + k0), *(const v8bf*)(a3 + k0 + 16));
    acc[0][0] = WMMA_BF16(A0, B0, acc[0][0]);
    acc[0][1] = WMMA_BF16(A0, B1, acc[0][1]);
    acc[1][0] = WMMA_BF16(A1, B0, acc[1][0]);
    acc[1][1] = WMMA_BF16(A1, B1, acc[1][1]);
    acc[2][0] = WMMA_BF16(A2, B0, acc[2][0]);
    acc[2][1] = WMMA_BF16(A2, B1, acc[2][1]);
    acc[3][0] = WMMA_BF16(A3, B0, acc[3][0]);
    acc[3][1] = WMMA_BF16(A3, B1, acc[3][1]);
  }
#pragma unroll
  for (int s = 0; s < 4; ++s) {
    const int o = mb + s * 16 + 8 * half;
#pragma unroll
    for (int j = 0; j < 2; ++j) {
      float* drow = D + (size_t)(n0 + j * 16 + lr) * 256 + o;
#pragma unroll
      for (int r = 0; r < 8; ++r) drow[r] = acc[s][j][r] + bias[o + r];
    }
  }
}

// ---------- top-down: fine[p][c] += coarse[p/2][c] ----------
__global__ void k_upadd(float* __restrict__ fine, const float* __restrict__ coarse,
                        int Wf, size_t n) {
  for (size_t i = blockIdx.x * (size_t)blockDim.x + threadIdx.x; i < n;
       i += (size_t)gridDim.x * blockDim.x) {
    size_t p = i >> 8; int c = (int)(i & 255);
    int y = (int)(p / Wf), xx = (int)(p % Wf);
    size_t pc = (size_t)(y >> 1) * (Wf >> 1) + (xx >> 1);
    fine[i] += coarse[pc * 256 + c];
  }
}

// ---------- AAC branch: implicit-GEMM 3x3 conv, 4M x 2N blocking, fused scale/shift+ReLU ----------
__global__ void k_gemm_aac(const __bf16* __restrict__ RW, const __bf16* __restrict__ X,
                           const float* __restrict__ sc, const float* __restrict__ sh,
                           __bf16* __restrict__ F, int H, int Wd) {
  const int lane = threadIdx.x & 31;
  const int half = lane >> 4;
  const int lr   = lane & 15;
  const int wave = threadIdx.x >> 5;
  const int n0 = blockIdx.x * 32;
  const int mb = wave * 64;
  const int p0 = n0 + lr;
  const int p1 = n0 + 16 + lr;
  const int y0 = p0 / Wd, x0 = p0 % Wd;
  const int y1 = p1 / Wd, x1 = p1 % Wd;
  v8f acc[4][2] = {};
  // hoisted weight bases (per-lane row, half offset baked in)
  const __bf16* a0 = RW + (size_t)(mb + lr) * 2304 + 8 * half;
  const __bf16* a1 = a0 + 16 * 2304;
  const __bf16* a2 = a0 + 32 * 2304;
  const __bf16* a3 = a0 + 48 * 2304;
  const __bf16* xh = X + 8 * half;
  for (int t = 0; t < 9; ++t) {                   // runtime tap loop: small code
    const int dy = t / 3 - 1, dx = t % 3 - 1;
    const int ya = y0 + dy, xa = x0 + dx;
    const int yb = y1 + dy, xb2 = x1 + dx;
    const bool ok0 = (ya >= 0) && (ya < H) && (xa >= 0) && (xa < Wd);
    const bool ok1 = (yb >= 0) && (yb < H) && (xb2 >= 0) && (xb2 < Wd);
    const __bf16* brow0 = xh + ((size_t)ya * Wd + xa) * 256;
    const __bf16* brow1 = xh + ((size_t)yb * Wd + xb2) * 256;
    const int toff = t * 256;
#pragma unroll
    for (int k0 = 0; k0 < 256; k0 += 32) {        // fully unrolled: immediate offsets
      v8bf b0lo = {}, b0hi = {}, b1lo = {}, b1hi = {};
      if (ok0) { b0lo = *(const v8bf*)(brow0 + k0); b0hi = *(const v8bf*)(brow0 + k0 + 16); }
      if (ok1) { b1lo = *(const v8bf*)(brow1 + k0); b1hi = *(const v8bf*)(brow1 + k0 + 16); }
      v16bf B0 = SHUF16(b0lo, b0hi);
      v16bf B1 = SHUF16(b1lo, b1hi);
      v16bf A0 = SHUF16(*(const v8bf*)(a0 + toff + k0), *(const v8bf*)(a0 + toff + k0 + 16));
      v16bf A1 = SHUF16(*(const v8bf*)(a1 + toff + k0), *(const v8bf*)(a1 + toff + k0 + 16));
      v16bf A2 = SHUF16(*(const v8bf*)(a2 + toff + k0), *(const v8bf*)(a2 + toff + k0 + 16));
      v16bf A3 = SHUF16(*(const v8bf*)(a3 + toff + k0), *(const v8bf*)(a3 + toff + k0 + 16));
      acc[0][0] = WMMA_BF16(A0, B0, acc[0][0]);
      acc[0][1] = WMMA_BF16(A0, B1, acc[0][1]);
      acc[1][0] = WMMA_BF16(A1, B0, acc[1][0]);
      acc[1][1] = WMMA_BF16(A1, B1, acc[1][1]);
      acc[2][0] = WMMA_BF16(A2, B0, acc[2][0]);
      acc[2][1] = WMMA_BF16(A2, B1, acc[2][1]);
      acc[3][0] = WMMA_BF16(A3, B0, acc[3][0]);
      acc[3][1] = WMMA_BF16(A3, B1, acc[3][1]);
    }
  }
#pragma unroll
  for (int s = 0; s < 4; ++s) {
    const int o = mb + s * 16 + 8 * half;
#pragma unroll
    for (int j = 0; j < 2; ++j) {
      __bf16* frow = F + (size_t)(n0 + j * 16 + lr) * 256 + o;
#pragma unroll
      for (int r = 0; r < 8; ++r)
        frow[r] = (__bf16)fmaxf(fmaf(acc[s][j][r], sc[o + r], sh[o + r]), 0.0f);
    }
  }
}

// ---------- SK squeeze: fea_s[c] = sum over 5 branches, mean over HW ----------
__global__ void k_reduce(const __bf16* __restrict__ F, float* __restrict__ S, int HW) {
  __shared__ float sm[256];
  const int c = blockIdx.x;
  float s = 0.0f;
  const size_t n = 5ull * HW;
  for (size_t i = threadIdx.x; i < n; i += 256)
    s += (float)F[i * 256 + c];
  sm[threadIdx.x] = s;
  __syncthreads();
  for (int d = 128; d > 0; d >>= 1) {
    if ((int)threadIdx.x < d) sm[threadIdx.x] += sm[threadIdx.x + d];
    __syncthreads();
  }
  if (threadIdx.x == 0) S[c] = sm[0] / (float)HW;
}

// ---------- FC -> per-branch logits -> softmax over 5 branches ----------
__global__ void k_attention(const float* __restrict__ feaS, const float* __restrict__ fw,
                            const float* __restrict__ fb, const float* __restrict__ sw,
                            const float* __restrict__ sb, float* __restrict__ att) {
  __shared__ float ss[256];
  __shared__ float sz[128];
  const int tid = threadIdx.x;
  ss[tid] = feaS[tid];
  __syncthreads();
  if (tid < 128) {
    float a = fb[tid];
    for (int c = 0; c < 256; ++c) a += ss[c] * fw[tid * 256 + c];
    sz[tid] = a;
  }
  __syncthreads();
  float lg[5];
  float mx = -1e30f;
#pragma unroll
  for (int r = 0; r < 5; ++r) {
    float a = sb[r * 256 + tid];
    const float* wrow = sw + ((size_t)r * 256 + tid) * 128;
    for (int d = 0; d < 128; ++d) a += sz[d] * wrow[d];
    lg[r] = a; mx = fmaxf(mx, a);
  }
  float den = 0.0f;
#pragma unroll
  for (int r = 0; r < 5; ++r) { lg[r] = expf(lg[r] - mx); den += lg[r]; }
#pragma unroll
  for (int r = 0; r < 5; ++r) att[r * 256 + tid] = lg[r] / den;
}

// ---------- out[c][p] = sum_r feas[r][p][c] * att[r][c] ----------
__global__ void k_combine(const __bf16* __restrict__ F, const float* __restrict__ att,
                          float* __restrict__ out, int HW) {
  size_t n = 256ull * HW;
  for (size_t i = blockIdx.x * (size_t)blockDim.x + threadIdx.x; i < n;
       i += (size_t)gridDim.x * blockDim.x) {
    int c = (int)(i / HW);
    size_t p = i % HW;
    float s = 0.0f;
#pragma unroll
    for (int r = 0; r < 5; ++r)
      s += (float)F[((size_t)r * HW + p) * 256 + c] * att[r * 256 + c];
    out[i] = s;
  }
}

extern "C" void kernel_launch(void* const* d_in, const int* in_sizes, int n_in,
                              void* d_out, int out_size, void* d_ws, size_t ws_size,
                              hipStream_t stream) {
  (void)in_sizes; (void)n_in; (void)out_size; (void)ws_size;
  const int Cin[4] = {256, 512, 1024, 2048};
  const int Wd[4]  = {128, 64, 32, 16};
  const int HWs[4] = {16384, 4096, 1024, 256};

  const float* x[4]  = {(const float*)d_in[0], (const float*)d_in[2],
                        (const float*)d_in[4], (const float*)d_in[6]};
  const float* lw[4] = {(const float*)d_in[1], (const float*)d_in[3],
                        (const float*)d_in[5], (const float*)d_in[7]};
  const float* lb   = (const float*)d_in[8];
  const float* aacw = (const float*)d_in[9];
  const float* aacb = (const float*)d_in[10];
  const float* bng  = (const float*)d_in[11];
  const float* bnb  = (const float*)d_in[12];
  const float* bnm  = (const float*)d_in[13];
  const float* bnv  = (const float*)d_in[14];
  const float* fcw  = (const float*)d_in[15];
  const float* fcb  = (const float*)d_in[16];
  const float* fcsw = (const float*)d_in[17];
  const float* fcsb = (const float*)d_in[18];

  char* ws = (char*)d_ws;
  size_t off = 0;
  auto take = [&](size_t bytes) -> char* {
    char* p = ws + off;
    off = (off + bytes + 255) & ~(size_t)255;
    return p;
  };
  __bf16* xb[4]; __bf16* lwb[4]; float* latf[4]; __bf16* latb[4]; __bf16* feas[4];
  for (int i = 0; i < 4; ++i) xb[i]  = (__bf16*)take((size_t)HWs[i] * Cin[i] * 2);
  for (int i = 0; i < 4; ++i) lwb[i] = (__bf16*)take((size_t)256 * Cin[i] * 2);
  __bf16* rw = (__bf16*)take(4ull * 5 * 256 * 2304 * 2);
  for (int i = 0; i < 4; ++i) latf[i] = (float*)take((size_t)HWs[i] * 256 * 4);
  for (int i = 0; i < 4; ++i) latb[i] = (__bf16*)take((size_t)HWs[i] * 256 * 2);
  for (int i = 0; i < 4; ++i) feas[i] = (__bf16*)take(5ull * HWs[i] * 256 * 2);
  float* feaS = (float*)take(4 * 256 * 4);
  float* att  = (float*)take(4 * 5 * 256 * 4);
  float* bnsc = (float*)take(4 * 5 * 256 * 4);
  float* bnsh = (float*)take(4 * 5 * 256 * 4);

  // 1) layout + precision conversions, weight prep
  for (int i = 0; i < 4; ++i) {
    dim3 gt(HWs[i] / 32, Cin[i] / 32);
    k_transpose_bf16<<<gt, dim3(256), 0, stream>>>(x[i], xb[i], Cin[i], HWs[i]);
    k_f32_to_bf16<<<dim3(512), dim3(256), 0, stream>>>(lw[i], lwb[i], (size_t)256 * Cin[i]);
  }
  k_rotate_w<<<dim3(4096), dim3(256), 0, stream>>>(aacw, rw);
  k_bnprep<<<dim3(20), dim3(256), 0, stream>>>(bng, bnb, bnm, bnv, aacb, bnsc, bnsh);

  // 2) lateral 1x1 convs (WMMA)
  for (int i = 0; i < 4; ++i)
    k_gemm_lat<<<dim3(HWs[i] / 32), dim3(128), 0, stream>>>(lwb[i], xb[i], lb + i * 256,
                                                            latf[i], Cin[i]);

  // 3) top-down pathway (ordered on stream)
  for (int i = 3; i >= 1; --i) {
    size_t n = (size_t)HWs[i - 1] * 256;
    k_upadd<<<dim3(2048), dim3(256), 0, stream>>>(latf[i - 1], latf[i], Wd[i - 1], n);
  }
  for (int i = 0; i < 4; ++i)
    k_f32_to_bf16<<<dim3(2048), dim3(256), 0, stream>>>(latf[i], latb[i],
                                                        (size_t)HWs[i] * 256);

  // 4) 5 rotated 3x3 convs per level (WMMA implicit GEMM), fused scale/shift + ReLU
  for (int i = 0; i < 4; ++i)
    for (int r = 0; r < 5; ++r) {
      const __bf16* RW = rw + (((size_t)i * 5 + r) * 256) * 2304;
      size_t bo = ((size_t)i * 5 + r) * 256;
      k_gemm_aac<<<dim3(HWs[i] / 32), dim3(128), 0, stream>>>(
          RW, latb[i], bnsc + bo, bnsh + bo,
          feas[i] + (size_t)r * HWs[i] * 256, Wd[i], Wd[i]);
    }

  // 5) SK squeeze + attention + combine
  for (int i = 0; i < 4; ++i)
    k_reduce<<<dim3(256), dim3(256), 0, stream>>>(feas[i], feaS + i * 256, HWs[i]);
  for (int i = 0; i < 4; ++i)
    k_attention<<<dim3(1), dim3(256), 0, stream>>>(feaS + i * 256,
                                                   fcw + (size_t)i * 128 * 256,
                                                   fcb + i * 128,
                                                   fcsw + (size_t)i * 5 * 256 * 128,
                                                   fcsb + (size_t)i * 5 * 256,
                                                   att + i * 5 * 256);
  float* outp = (float*)d_out;
  size_t oacc = 0;
  for (int i = 0; i < 4; ++i) {
    k_combine<<<dim3(2048), dim3(256), 0, stream>>>(feas[i], att + i * 5 * 256,
                                                    outp + oacc, HWs[i]);
    oacc += (size_t)256 * HWs[i];
  }
}
